// SelfAttention_12678743457848
// MI455X (gfx1250) — matrix operand order repaired and checked by reading, not compiled, not run on hardware
//
#include <hip/hip_runtime.h>

#define B_  8
#define S_  2048
#define DM  1024
#define DH  64

typedef __attribute__((ext_vector_type(16))) __bf16 v16bf;
typedef __attribute__((ext_vector_type(8)))  __bf16 v8bf;
typedef __attribute__((ext_vector_type(4)))  __bf16 v4bf;
typedef __attribute__((ext_vector_type(2)))  __bf16 v2bf;
typedef __attribute__((ext_vector_type(8)))  float  v8f;
typedef __attribute__((ext_vector_type(4)))  float  v4f;

// ---- helpers -------------------------------------------------------------

// streaming (non-temporal) f32 -> bf16 x8
__device__ __forceinline__ v8bf cvt8nt(const float* p) {
    v4f a = __builtin_nontemporal_load((const v4f*)p);
    v4f b = __builtin_nontemporal_load((const v4f*)(p + 4));
    v8bf r;
#pragma unroll
    for (int i = 0; i < 4; ++i) { r[i] = (__bf16)a[i]; r[i + 4] = (__bf16)b[i]; }
    return r;
}

__device__ __forceinline__ v8bf ld8bf(const __bf16* p) { return *(const v8bf*)(p); }

__device__ __forceinline__ v16bf combine(v8bf lo, v8bf hi) {
    return __builtin_shufflevector(lo, hi, 0,1,2,3,4,5,6,7,8,9,10,11,12,13,14,15);
}

__device__ __forceinline__ v8f wmma_bf16(v16bf a, v16bf b, v8f c) {
    return __builtin_amdgcn_wmma_f32_16x16x32_bf16(
        false, a, false, b, (short)0, c, false, false);
}

// ---- kernel 0: convert the three 64x1024 weight matrices to bf16 ---------

__global__ __launch_bounds__(256) void wcvt_kernel(
    const float* __restrict__ Wq, const float* __restrict__ Wk,
    const float* __restrict__ Wv, __bf16* __restrict__ out)
{
    const int i = (blockIdx.x * 256 + threadIdx.x) * 4;   // 0 .. 196604
    const float* src; int j;
    if (i < 65536)       { src = Wq; j = i; }
    else if (i < 131072) { src = Wk; j = i - 65536; }
    else                 { src = Wv; j = i - 131072; }
    v4f x = *(const v4f*)(src + j);
    v4bf o;
#pragma unroll
    for (int e = 0; e < 4; ++e) o[e] = (__bf16)x[e];
    *(v4bf*)(out + i) = o;
}

// ---- kernel 1: projection GEMM  y = X @ W^T ------------------------------
// transposed==0 : out[row*64 + h]        (row-major [B,S,64], bf16)
// transposed==1 : out[(b*64+h)*S + s]    (V^T layout [B,64,S], bf16)

__global__ __launch_bounds__(128) void proj_kernel(
    const float* __restrict__ X, const __bf16* __restrict__ Wb,
    __bf16* __restrict__ out, int transposed)
{
    const int wave = threadIdx.x >> 5;
    const int lane = threadIdx.x & 31;
    const int l    = lane & 15;
    const int hf   = lane >> 4;
    const int m0   = (blockIdx.x * 4 + wave) * 16;    // row-tile base in [0,B*S)

    v8f acc[4] = {};
    const float* xrow = X + (size_t)(m0 + l) * DM;

    for (int k = 0; k < DM; k += 32) {
        // A fragment (rows of X, 16x32): K chunks {8hf..} and {16+8hf..}
        v16bf afrag = combine(cvt8nt(xrow + k + 8 * hf),
                              cvt8nt(xrow + k + 16 + 8 * hf));
        // All four B fragments first, then four independent WMMAs
        v16bf bfrag[4];
#pragma unroll
        for (int t = 0; t < 4; ++t) {
            const __bf16* wr = Wb + (size_t)(t * 16 + l) * DM + k + 8 * hf;
            bfrag[t] = combine(ld8bf(wr), ld8bf(wr + 16));
        }
#pragma unroll
        for (int t = 0; t < 4; ++t)
            acc[t] = wmma_bf16(afrag, bfrag[t], acc[t]);
    }

    if (!transposed) {
#pragma unroll
        for (int t = 0; t < 4; ++t)
#pragma unroll
            for (int r = 0; r < 8; ++r)
                out[(size_t)(m0 + r + 8 * hf) * DH + t * 16 + l] = (__bf16)acc[t][r];
    } else {
        const int b  = m0 / S_;
        const int s0 = m0 % S_;
#pragma unroll
        for (int t = 0; t < 4; ++t) {
            v8bf pk;
#pragma unroll
            for (int r = 0; r < 8; ++r) pk[r] = (__bf16)acc[t][r];
            *(v8bf*)(out + (size_t)(b * DH + t * 16 + l) * S_ + s0 + 8 * hf) = pk;
        }
    }
}

// ---- kernel 2: causal flash attention, transposed formulation ------------

__device__ __forceinline__ void attn_tile(
    int b, int q0, int l, int hf,
    const __bf16* __restrict__ Qb, const __bf16* __restrict__ Kb,
    const __bf16* __restrict__ VT, float* __restrict__ O)
{
    // scores computed in exp2 domain: SC = (1/sqrt(64)) * log2(e)
    const float SC = 0.18033688011112042f;

    const __bf16* qrow = Qb + ((size_t)b * S_ + q0 + l) * DH;
    const v16bf bQ0 = combine(ld8bf(qrow + 8 * hf),      ld8bf(qrow + 16 + 8 * hf));
    const v16bf bQ1 = combine(ld8bf(qrow + 32 + 8 * hf), ld8bf(qrow + 48 + 8 * hf));

    float m    = -1e30f;
    float lsum = 0.0f;
    v8f   acc[4] = {};
    const int qi   = q0 + l;
    const int kend = q0 + 16;

    for (int kb = 0; kb < kend; kb += 32) {
        const __bf16* kr0 = Kb + ((size_t)b * S_ + kb + l) * DH;
        const __bf16* kr1 = kr0 + 16 * DH;
        if (kb + 32 < kend) {                      // prefetch next K tile
            __builtin_prefetch(kr0 + 32 * DH, 0, 1);
            __builtin_prefetch(kr1 + 32 * DH, 0, 1);
        }
        v16bf aK00 = combine(ld8bf(kr0 + 8 * hf),      ld8bf(kr0 + 16 + 8 * hf));
        v16bf aK01 = combine(ld8bf(kr0 + 32 + 8 * hf), ld8bf(kr0 + 48 + 8 * hf));
        v16bf aK10 = combine(ld8bf(kr1 + 8 * hf),      ld8bf(kr1 + 16 + 8 * hf));
        v16bf aK11 = combine(ld8bf(kr1 + 32 + 8 * hf), ld8bf(kr1 + 48 + 8 * hf));

        v8f t0 = {}; v8f t1 = {};
        t0 = wmma_bf16(aK00, bQ0, t0);             // interleaved independent chains
        t1 = wmma_bf16(aK10, bQ0, t1);
        t0 = wmma_bf16(aK01, bQ1, t0);
        t1 = wmma_bf16(aK11, bQ1, t1);

        // scale (log2 domain) + causal mask; element (key = kb+r+8hf (+16), q = qi)
        float s0v[8], s1v[8];
        float tmax = -1e30f;
#pragma unroll
        for (int r = 0; r < 8; ++r) {
            int k0 = kb + r + 8 * hf;
            float x0 = (k0      <= qi) ? t0[r] * SC : -1e30f;
            float x1 = (k0 + 16 <= qi) ? t1[r] * SC : -1e30f;
            s0v[r] = x0; s1v[r] = x1;
            tmax = fmaxf(tmax, fmaxf(x0, x1));
        }
        tmax = fmaxf(tmax, __shfl_xor(tmax, 16, 32));

        const float mnew  = fmaxf(m, tmax);
        const float alpha = __builtin_amdgcn_exp2f(m - mnew);
        float rsum = 0.0f;
#pragma unroll
        for (int r = 0; r < 8; ++r) {
            s0v[r] = __builtin_amdgcn_exp2f(s0v[r] - mnew);
            s1v[r] = __builtin_amdgcn_exp2f(s1v[r] - mnew);
            rsum  += s0v[r] + s1v[r];
        }
        rsum += __shfl_xor(rsum, 16, 32);
        lsum  = lsum * alpha + rsum;
        m     = mnew;
#pragma unroll
        for (int t = 0; t < 4; ++t)
#pragma unroll
            for (int r = 0; r < 8; ++r) acc[t][r] *= alpha;

        // P^T B-fragment from packed bf16 pairs (half the shuffles/cvts)
        unsigned pk0[4], pk1[4];
#pragma unroll
        for (int i = 0; i < 4; ++i) {
            v2bf a; a[0] = (__bf16)s0v[2 * i]; a[1] = (__bf16)s0v[2 * i + 1];
            v2bf c; c[0] = (__bf16)s1v[2 * i]; c[1] = (__bf16)s1v[2 * i + 1];
            pk0[i] = __builtin_bit_cast(unsigned, a);
            pk1[i] = __builtin_bit_cast(unsigned, c);
        }
        union { v16bf v; unsigned u[8]; } bp;
#pragma unroll
        for (int i = 0; i < 4; ++i) {
            unsigned o0 = (unsigned)__shfl_xor((int)pk0[i], 16, 32);
            unsigned o1 = (unsigned)__shfl_xor((int)pk1[i], 16, 32);
            bp.u[i]     = hf ? o1     : pk0[i];   // keys kb+0..7  / kb+16..23
            bp.u[i + 4] = hf ? pk1[i] : o0;       // keys kb+8..15 / kb+24..31
        }

        // O^T += V^T(16h x 32keys) * P^T ; V^T rows contiguous in key
        const __bf16* vbase = VT + (size_t)b * DH * S_ + kb;
#pragma unroll
        for (int t = 0; t < 4; ++t) {
            const __bf16* vrow = vbase + (size_t)(t * 16 + l) * S_;
            v16bf aV = combine(ld8bf(vrow + 16 * hf), ld8bf(vrow + 16 * hf + 8));
            acc[t] = wmma_bf16(aV, bp.v, acc[t]);
        }
    }

    const float inv = 1.0f / lsum;
    float* orow = O + ((size_t)b * S_ + q0 + l) * DH;
#pragma unroll
    for (int t = 0; t < 4; ++t) {
        v4f o0, o1;
#pragma unroll
        for (int r = 0; r < 4; ++r) { o0[r] = acc[t][r] * inv; o1[r] = acc[t][r + 4] * inv; }
        __builtin_nontemporal_store(o0, (v4f*)(orow + t * 16 + 8 * hf));
        __builtin_nontemporal_store(o1, (v4f*)(orow + t * 16 + 8 * hf + 4));
    }
}

// Triangle-balanced: each wave handles q-tiles (j, NT-1-j) -> uniform work.
__global__ __launch_bounds__(128) void attn_kernel(
    const __bf16* __restrict__ Qb, const __bf16* __restrict__ Kb,
    const __bf16* __restrict__ VT, float* __restrict__ O)
{
    const int wave = threadIdx.x >> 5;
    const int lane = threadIdx.x & 31;
    const int l    = lane & 15;
    const int hf   = lane >> 4;
    const int w    = blockIdx.x * 4 + wave;     // 0 .. B*(S/16)/2 - 1
    const int b    = w / (S_ / 32);             // S/32 = 64 tile-pairs per batch
    const int j    = w % (S_ / 32);

    attn_tile(b, j * 16, l, hf, Qb, Kb, VT, O);
    attn_tile(b, (S_ / 16 - 1 - j) * 16, l, hf, Qb, Kb, VT, O);
}

// ---- host launcher -------------------------------------------------------

extern "C" void kernel_launch(void* const* d_in, const int* in_sizes, int n_in,
                              void* d_out, int out_size, void* d_ws, size_t ws_size,
                              hipStream_t stream) {
    (void)in_sizes; (void)n_in; (void)out_size; (void)ws_size;
    const float* Qin = (const float*)d_in[0];
    const float* Kin = (const float*)d_in[1];
    const float* Vin = (const float*)d_in[2];
    const float* Wq  = (const float*)d_in[3];
    const float* Wk  = (const float*)d_in[4];
    const float* Wv  = (const float*)d_in[5];

    __bf16* qb   = (__bf16*)d_ws;                     // [B,S,64]  2 MB
    __bf16* kbuf = qb + (size_t)B_ * S_ * DH;         // [B,S,64]  2 MB
    __bf16* vt   = kbuf + (size_t)B_ * S_ * DH;       // [B,64,S]  2 MB
    __bf16* wb   = vt + (size_t)B_ * S_ * DH;         // 3x[64,1024] 384 KB
    __bf16* wbq  = wb;
    __bf16* wbk  = wb + (size_t)DH * DM;
    __bf16* wbv  = wb + (size_t)2 * DH * DM;

    wcvt_kernel<<<(3 * DH * DM) / (256 * 4), 256, 0, stream>>>(Wq, Wk, Wv, wb);

    dim3 blk(128);
    const int mblocks = (B_ * S_) / 64;               // 256 blocks, 4 waves each
    proj_kernel<<<mblocks, blk, 0, stream>>>(Qin, wbq, qb, 0);
    proj_kernel<<<mblocks, blk, 0, stream>>>(Kin, wbk, kbuf, 0);
    proj_kernel<<<mblocks, blk, 0, stream>>>(Vin, wbv, vt, 1);

    const int pairs = B_ * (S_ / 32);                 // 512 waves, uniform work
    attn_kernel<<<pairs / 4, blk, 0, stream>>>(qb, kbuf, vt, (float*)d_out);
}
